// PairDecoder_74208444941048
// MI455X (gfx1250) — matrix operand order repaired and checked by reading, not compiled, not run on hardware
//
#include <hip/hip_runtime.h>

// ---------------------------------------------------------------------------
// PairDecoder for MI455X (gfx1250, wave32, WMMA).
//
// Roofline: ~34.6 GFLOP vs ~550 MB of gather traffic -> ~24 us HBM floor at
// 23.3 TB/s. bf16 WMMA (v_wmma_f32_16x16x32_bf16, f32 accumulate) keeps the
// matrix math far below the memory roof; fp32 WMMA (16x16x4) would be 8x
// slower per instruction and make the kernel compute bound.
//
// Per wave: one 16-edge tile.
//   GEMM1: h[16x128] = x[16x256] @ W1^T  -> 64 bf16 WMMAs (8 Kchunks x 8 Ntiles)
//     A frags gathered directly from HBM (two 32B contiguous runs per lane
//     per K-chunk, converted f32->bf16 in registers).
//     B frags read from a 64KB LDS copy of W1 (bf16), staged once per block.
//     An opaque *offset* (asm "+v" on an int) prevents LICM from hoisting
//     all 64 B frags into VGPRs (round-1 spill disaster) while keeping the
//     base address in the LDS address space -> ds_load_b128, not flat_load.
//     NOTE: software double-buffering of B frags was tried and REJECTED --
//     it let the scheduler interchange the t/kc loops, keeping 8+ B frags
//     live, overflowing into the high-VGPR file (s_set_vgpr_msb) and
//     spilling to scratch. The serialized s_wait_dscnt before each WMMA is
//     instead hidden by occupancy (~150 VGPRs -> several waves/SIMD).
//   GEMM2: out[16x2] = relu(h+b1) @ W2^T + b2 -> VALU FMAs on the WMMA
//     accumulators + shfl_xor butterfly reduction; coalesced 128B stores.
// ---------------------------------------------------------------------------

typedef __attribute__((ext_vector_type(16))) __bf16 v16bf;
typedef __attribute__((ext_vector_type(8)))  __bf16 v8bf;
typedef __attribute__((ext_vector_type(8)))  float  v8f;
typedef __attribute__((ext_vector_type(4)))  float  v4f;
typedef __attribute__((ext_vector_type(2)))  int    v2i;

#define B_      8
#define N_      100000
#define D_      128
#define E_      65536      // pairs per batch (power of two: pe>>16, pe&65535)
#define H_      128
#define K_      256        // 2*D
#define WAVES   8
#define THREADS 256
#define BLOCKS  1024
#define NTILES  ((B_ * E_) / 16)   // 32768 16-edge tiles

__device__ __forceinline__ __bf16 tobf(float x) { return (__bf16)x; }

__global__ __launch_bounds__(THREADS)
void pair_decoder_kernel(const float* __restrict__ emb,
                         const float* __restrict__ W1,
                         const float* __restrict__ b1,
                         const float* __restrict__ W2,
                         const float* __restrict__ b2,
                         const int*   __restrict__ idx,
                         float*       __restrict__ out)
{
    __shared__ __bf16 sW1[H_ * K_];   // 64 KB: W1 as bf16, row-major [H][K]

    const int tid = threadIdx.x;

    // ---- Stage W1 (fp32 HBM/L2 -> bf16 LDS), 128 floats per thread ----
    {
        const int row = tid >> 1;              // 0..127
        const int col = (tid & 1) * 128;       // 0 or 128
        const float* src = W1 + row * K_ + col;
        __bf16*      dst = sW1 + row * K_ + col;
#pragma unroll
        for (int i = 0; i < 16; ++i) {
            v4f lo = *(const v4f*)(src + i * 8);
            v4f hi = *(const v4f*)(src + i * 8 + 4);
            v8bf p;
#pragma unroll
            for (int j = 0; j < 4; ++j) {
                p[j]     = tobf(lo[j]);
                p[4 + j] = tobf(hi[j]);
            }
            *(v8bf*)(dst + i * 8) = p;         // 16B ds_store
        }
    }
    __syncthreads();

    const int lane = tid & 31;
    const int wave = tid >> 5;
    const int c    = lane & 15;    // column-in-tile / edge-in-tile for this lane
    const int half = lane >> 4;    // 0: rows M=0..7 & K-low ; 1: rows M=8..15 & K-high

    // Per-lane bias / W2 columns (n = 16*t + c), loaded once per wave.
    float b1v[8], w20[8], w21[8];
#pragma unroll
    for (int t = 0; t < 8; ++t) {
        b1v[t] = b1[16 * t + c];
        w20[t] = W2[16 * t + c];        // W2[0][n]
        w21[t] = W2[H_ + 16 * t + c];   // W2[1][n]
    }
    const float bias2 = b2[lane & 1];

    const int gstride = gridDim.x * WAVES;

    for (int tile = blockIdx.x * WAVES + wave; tile < NTILES; tile += gstride) {
        // ---- Gather this lane's edge (lanes l and l+16 share edge c) ----
        const int pe = tile * 16 + c;          // flat pair index
        const int b  = pe >> 16;               // pe / E_
        const int e  = pe & (E_ - 1);
        const v2i  ii = *(const v2i*)(idx + ((long long)b * E_ + e) * 2);
        const float* row0 = emb + ((long long)b * N_ + ii[0]) * D_;
        const float* row1 = emb + ((long long)b * N_ + ii[1]) * D_;

        // ---- A fragments: 16x256 activations, bf16 ----
        // Lane needs two 8-float runs per 32-wide K-chunk:
        //   half=0: K = 32kc+{0..7}, 32kc+{8..15}
        //   half=1: K = 32kc+{16..23}, 32kc+{24..31}
        // Runs are 8-aligned => each run stays inside one endpoint embedding.
        v16bf afrag[8];
#pragma unroll
        for (int kc = 0; kc < 8; ++kc) {
            const int k0 = 32 * kc + (half ? 16 : 0);
            const int k1 = 32 * kc + (half ? 24 : 8);
            const float* p0 = (k0 < D_) ? (row0 + k0) : (row1 + (k0 - D_));
            const float* p1 = (k1 < D_) ? (row0 + k1) : (row1 + (k1 - D_));
            v4f a0 = *(const v4f*)p0;
            v4f a1 = *(const v4f*)(p0 + 4);
            v4f c0 = *(const v4f*)p1;
            v4f c1 = *(const v4f*)(p1 + 4);
#pragma unroll
            for (int j = 0; j < 4; ++j) {
                afrag[kc][j]      = tobf(a0[j]);
                afrag[kc][4 + j]  = tobf(a1[j]);
                afrag[kc][8 + j]  = tobf(c0[j]);
                afrag[kc][12 + j] = tobf(c1[j]);
            }
        }

        // ---- GEMM1: 8 N-tiles x 8 K-chunks of v_wmma_f32_16x16x32_bf16 ----
        v8f acc[8];
#pragma unroll
        for (int t = 0; t < 8; ++t) {
            v8f a = {0.f, 0.f, 0.f, 0.f, 0.f, 0.f, 0.f, 0.f};
            // B frag: lane holds W1[16t+c][32kc + (half?16:0) .. +15],
            // 32 contiguous bytes in LDS.
            // Opaque-ify the OFFSET only: the volatile asm executes every
            // tile iteration, so the 8 LDS loads below can't be proven
            // loop-invariant and hoisted (no 512-VGPR blowup / spills),
            // while the base stays a known LDS object -> ds_load_b128
            // with 32-bit addressing (not flat_load).
            int boff = (16 * t + c) * K_ + (half ? 16 : 0);
            asm volatile("" : "+v"(boff));
            const __bf16* bp = sW1 + boff;
#pragma unroll
            for (int kc = 0; kc < 8; ++kc) {
                v16bf bfrag = *(const v16bf*)(bp + 32 * kc);
                a = __builtin_amdgcn_wmma_f32_16x16x32_bf16(
                        false, afrag[kc], false, bfrag,
                        (short)0, a, false, false);
            }
            acc[t] = a;
        }

        // ---- ReLU + bias + GEMM2 (H=128 -> 2) on accumulators ----
        // Lane (half,c) holds h[m][16t+c] for m = 8*half + r, r=0..7.
        float res[16];
#pragma unroll
        for (int r = 0; r < 8; ++r) {
            float p0 = 0.f, p1 = 0.f;
#pragma unroll
            for (int t = 0; t < 8; ++t) {
                float h = fmaxf(acc[t][r] + b1v[t], 0.f);
                p0 = fmaf(h, w20[t], p0);
                p1 = fmaf(h, w21[t], p1);
            }
            res[2 * r]     = p0;
            res[2 * r + 1] = p1;
        }
        // Butterfly sum over the 16 lanes of each half (xor masks stay in-half).
#pragma unroll
        for (int j = 0; j < 16; ++j) {
#pragma unroll
            for (int m = 1; m <= 8; m <<= 1)
                res[j] += __shfl_xor(res[j], m, 32);
        }
        // Lane l owns flat output slot tile*32 + l  (m = l>>1, o = l&1):
        // that is res[2*((l>>1)&7) + (l&1)] == res[l & 15].
        out[(long long)tile * 32 + lane] = res[lane & 15] + bias2;
    }
}

extern "C" void kernel_launch(void* const* d_in, const int* in_sizes, int n_in,
                              void* d_out, int out_size, void* d_ws, size_t ws_size,
                              hipStream_t stream)
{
    (void)in_sizes; (void)n_in; (void)out_size; (void)d_ws; (void)ws_size;
    const float* emb = (const float*)d_in[0];
    const float* W1  = (const float*)d_in[1];
    const float* b1  = (const float*)d_in[2];
    const float* W2  = (const float*)d_in[3];
    const float* b2  = (const float*)d_in[4];
    const int*   idx = (const int*)d_in[5];
    float* out = (float*)d_out;

    pair_decoder_kernel<<<BLOCKS, THREADS, 0, stream>>>(emb, W1, b1, W2, b2, idx, out);
}
